// Attention_14276471292256
// MI455X (gfx1250) — matrix operand order repaired and checked by reading, not compiled, not run on hardware
//
#include <hip/hip_runtime.h>
#include <hip/hip_bf16.h>
#include <stdint.h>

// MHA block for B=4, S=2048, D=2048, H=16, hd=128 on gfx1250 (MI455X).
// bf16 WMMA (v_wmma_f32_16x16x32_bf16) for all matmuls, f32 accumulate.
// TDM (tensor_load_to_lds) stages the non-transposed tiles (GEMM A, attn K).
// Workspace: 192 MiB (see kernel_launch).

typedef __attribute__((ext_vector_type(16))) __bf16 v16bf;
typedef __attribute__((ext_vector_type(8)))  __bf16 v8bf;
typedef __attribute__((ext_vector_type(8)))  float  v8f;
typedef __attribute__((ext_vector_type(4)))  unsigned int u32x4;
typedef __attribute__((ext_vector_type(8)))  int i32x8;
typedef __attribute__((ext_vector_type(4)))  int i32x4;

#define DEV __device__ __forceinline__

DEV __bf16 f2bf(float f) {
  uint32_t u = __builtin_bit_cast(uint32_t, f);
  u += 0x7FFFu + ((u >> 16) & 1u);            // round-to-nearest-even
  uint16_t h = (uint16_t)(u >> 16);
  return __builtin_bit_cast(__bf16, h);
}
DEV float bf2f(__bf16 b) {
  uint16_t h = __builtin_bit_cast(uint16_t, b);
  uint32_t u = ((uint32_t)h) << 16;
  return __builtin_bit_cast(float, u);
}
DEV v16bf load16(const __bf16* p0, const __bf16* p1) {
  union { v16bf v; v8bf h[2]; } u;
  u.h[0] = *(const v8bf*)p0;
  u.h[1] = *(const v8bf*)p1;
  return u.v;
}
DEV v8f wmma_bf16(v16bf a, v16bf b, v8f c) {
  return __builtin_amdgcn_wmma_f32_16x16x32_bf16(false, a, false, b, (short)0, c,
                                                 false, false);
}

// Flat LDS pointer -> wave-relative LDS byte offset (ISA 10.2: addr[31:0]).
DEV uint32_t lds_off(const void* p) { return (uint32_t)(uintptr_t)p; }

// Issue a 2D TDM load: tile_y rows of tile_x bf16 elements, global row stride
// row_stride elements, packed contiguously into LDS at lds_addr.
// Descriptor layout per CDNA5 ISA.
// This toolchain exposes the 6-arg builtin (clang-23 / therock-10.0 form).
DEV void tdm_load_2d(uint32_t lds_addr, const void* gptr,
                     uint32_t tile_x, uint32_t tile_y, uint32_t row_stride) {
  uint64_t ga = (uint64_t)(uintptr_t)gptr;
  u32x4 g0;
  g0[0] = 1u;                                            // count=1, no gather
  g0[1] = lds_addr;                                      // lds_addr (bytes)
  g0[2] = (uint32_t)ga;                                  // global_addr[31:0]
  g0[3] = (uint32_t)((ga >> 32) & 0x1FFFFFFu) | (2u << 30); // [56:32] | type=2
  i32x8 g1;
  g1[0] = (int)(1u << 16);                               // data_size=1 (2B)
  g1[1] = (int)((tile_x & 0xFFFFu) << 16);               // tensor_dim0 lo
  g1[2] = (int)((tile_x >> 16) | ((tile_y & 0xFFFFu) << 16)); // d0 hi|d1 lo
  g1[3] = (int)((tile_y >> 16) | ((tile_x & 0xFFFFu) << 16)); // d1 hi|tile_dim0
  g1[4] = (int)(tile_y & 0xFFFFu);                       // tile_dim1, tile_dim2=0
  g1[5] = (int)row_stride;                               // dim0_stride[31:0]
  g1[6] = 0;                                             // d0s hi | d1s lo
  g1[7] = 0;
  i32x4 g2; g2[0] = 1; g2[1] = 1; g2[2] = 0; g2[3] = 0;  // dim2=1, dim3=1
  i32x4 g3; g3[0] = 0; g3[1] = (int)(1u << 16); g3[2] = 0; g3[3] = 0; // dim4=1
  i32x8 g4 = {0, 0, 0, 0, 0, 0, 0, 0};                   // unused trailing group
  __builtin_amdgcn_tensor_load_to_lds(g0, g1, g2, g3, g4, 0);
}
DEV void wait_tensor0() { __builtin_amdgcn_s_wait_tensorcnt(0); }

// ---------------------------------------------------------------- convert
__global__ void cvt_f32_to_bf16(const float* __restrict__ in,
                                __bf16* __restrict__ out, long n4) {
  long i = (long)blockIdx.x * blockDim.x + threadIdx.x;
  if (i >= n4) return;
  float4 f = ((const float4*)in)[i];
  __bf16* o = out + i * 4;
  o[0] = f2bf(f.x); o[1] = f2bf(f.y); o[2] = f2bf(f.z); o[3] = f2bf(f.w);
}

// ---------------------------------------------------------------- GEMM
// C[M,N] = A[M,K] x W[K,N], row-major. Block tile 128x128, 4 waves, each wave
// 64x64 as 4x4 WMMA frags, K-step 32. A-tiles staged by TDM; W-tiles staged
// transposed (Bs[n][k]) by the waves so B-frags are contiguous 16B loads.
template <bool OUT_BF16>
__global__ __launch_bounds__(128) void gemm_bf16_k(
    const __bf16* __restrict__ A, const __bf16* __restrict__ W,
    void* __restrict__ C, int M, int N, int K) {
  constexpr int TM = 128, TN = 128, TK = 32;
  __shared__ __bf16 As[2][TM][TK];   // 16 KB (TDM destination)
  __shared__ __bf16 Bs[2][TN][TK];   // 16 KB
  const int tid = threadIdx.x, lane = tid & 31, wave = tid >> 5;
  const int wy = wave >> 1, wx = wave & 1;
  const long gm0 = (long)blockIdx.y * TM, gn0 = (long)blockIdx.x * TN;
  const int m = lane & 15;
  const int koff = (lane >> 4) << 4;
  const int kb = (lane >> 4) << 4;
  const int rbase = (lane >> 4) << 3;

  v8f acc[4][4] = {};

  auto stageA = [&](int buf, long k0) {
    if (wave == 0)   // wave-uniform descriptor; tensor ops issue once per wave
      tdm_load_2d(lds_off(&As[buf][0][0]), &A[gm0 * K + k0], TK, TM, K);
  };
  auto stageW = [&](int buf, long k0) {
#pragma unroll
    for (int c = 0; c < 4; ++c) {
      int id = tid * 4 + c;                    // 0..511
      int kk = id >> 4, n8 = (id & 15) << 3;   // 32 k-rows x 16 chunks
      v8bf t = *(const v8bf*)&W[(k0 + kk) * (long)N + gn0 + n8];
#pragma unroll
      for (int j = 0; j < 8; ++j) Bs[buf][n8 + j][kk] = t[j];
    }
  };

  stageA(0, 0);
  stageW(0, 0);
  wait_tensor0();
  __syncthreads();
  const int ks_end = K / TK;
  for (int ks = 0; ks < ks_end; ++ks) {
    int buf = ks & 1;
    if (ks + 1 < ks_end) {
      stageA(buf ^ 1, (long)(ks + 1) * TK);    // DMA overlaps the WMMAs below
      stageW(buf ^ 1, (long)(ks + 1) * TK);
    }
    if (ks + 2 < ks_end)
      __builtin_prefetch(&W[((long)(ks + 2) * TK + (tid & 31)) * N + gn0], 0, 3);

    v16bf af[4];
#pragma unroll
    for (int i = 0; i < 4; ++i) {
      const __bf16* p = &As[buf][wy * 64 + i * 16 + m][koff];
      af[i] = load16(p, p + 8);
    }
#pragma unroll
    for (int j = 0; j < 4; ++j) {              // j-outer: each B-frag loaded once
      const __bf16* p = &Bs[buf][wx * 64 + j * 16 + m][kb];
      v16bf bf = load16(p, p + 8);
#pragma unroll
      for (int i = 0; i < 4; ++i) acc[i][j] = wmma_bf16(af[i], bf, acc[i][j]);
    }
    wait_tensor0();
    __syncthreads();
  }

#pragma unroll
  for (int i = 0; i < 4; ++i)
#pragma unroll
    for (int j = 0; j < 4; ++j) {
      long col = gn0 + wx * 64 + j * 16 + m;
#pragma unroll
      for (int r = 0; r < 8; ++r) {
        long row = gm0 + wy * 64 + i * 16 + r + rbase;
        float v = acc[i][j][r];
        if (OUT_BF16) ((__bf16*)C)[row * N + col] = f2bf(v);
        else          ((float*)C)[row * N + col] = v;
      }
    }
}

// ---------------------------------------------------------------- RoPE
__global__ void rope_k(__bf16* __restrict__ q, __bf16* __restrict__ k,
                       const float* __restrict__ fc, const float* __restrict__ fs) {
  long idx = (long)blockIdx.x * blockDim.x + threadIdx.x;  // 4*2048*16*64
  int p = (int)(idx & 63);
  long t = idx >> 6;
  int h = (int)(t & 15);
  long bs = t >> 4;
  int s = (int)(bs & 2047);
  float c = fc[s * 64 + p], sn = fs[s * 64 + p];
  long base = bs * 2048 + h * 128 + p * 2;
  {
    float xr = bf2f(q[base]), xi = bf2f(q[base + 1]);
    q[base] = f2bf(xr * c - xi * sn);
    q[base + 1] = f2bf(xr * sn + xi * c);
  }
  {
    float xr = bf2f(k[base]), xi = bf2f(k[base + 1]);
    k[base] = f2bf(xr * c - xi * sn);
    k[base + 1] = f2bf(xr * sn + xi * c);
  }
}

// ---------------------------------------------------------------- flash attn
// Grid: (B*H, S/128). Block: 256 threads = 8 waves; each wave owns 16 q-rows.
// K key-blocks staged by TDM; V staged transposed; P routed through LDS.
__global__ __launch_bounds__(256) void flash_attn_k(
    const __bf16* __restrict__ Q, const __bf16* __restrict__ Kmat,
    const __bf16* __restrict__ V, const float* __restrict__ mask,
    __bf16* __restrict__ O) {
  constexpr int S = 2048, D = 2048, HD = 128, NB = 64;
  __shared__ __bf16 Ks[NB][HD];      // 16 KB, [key][dim]: TDM destination
  __shared__ __bf16 Vs[HD][NB];      // 16 KB, [dim][key]
  __shared__ __bf16 Ps[8][16][NB];   // 16 KB, per-wave P tiles

  const int bh = blockIdx.x;
  const int b = bh >> 4, h = bh & 15;
  const int tid = threadIdx.x, lane = tid & 31, wave = tid >> 5;
  const long qbase = (long)blockIdx.y * 128 + wave * 16;
  const long base_bh = ((long)b * S) * D + (long)h * HD;

  const int m = lane & 15;
  const int koff = (lane >> 4) << 4;
  const int kb = (lane >> 4) << 4;
  const int rbase = (lane >> 4) << 3;

  v16bf qf[4];
  {
    const __bf16* qr = Q + base_bh + (qbase + m) * (long)D;
#pragma unroll
    for (int kk = 0; kk < 4; ++kk) {
      const __bf16* p = qr + kk * 32 + koff;
      qf[kk] = load16(p, p + 8);
    }
  }

  v8f o[8] = {};
  float mrow[8], lsum[8];
#pragma unroll
  for (int r = 0; r < 8; ++r) { mrow[r] = -__builtin_inff(); lsum[r] = 0.f; }
  const float scale = 0.088388347648318447f;  // 1/sqrt(128)

  for (int nb = 0; nb < S; nb += NB) {
    __syncthreads();  // previous iteration's LDS reads complete
    if (wave == 0)    // 64x128 bf16 tile, row stride D
      tdm_load_2d(lds_off(&Ks[0][0]), &Kmat[base_bh + (long)nb * D], HD, NB, D);
#pragma unroll
    for (int c = 0; c < 4; ++c) {              // V staged transposed
      int id = tid * 4 + c;                    // 0..1023
      int key = id >> 4, d8 = (id & 15) << 3;
      v8bf tv = *(const v8bf*)&V[base_bh + (long)(nb + key) * D + d8];
#pragma unroll
      for (int j = 0; j < 8; ++j) Vs[d8 + j][key] = tv[j];
    }
    if (nb + NB < S)
      __builtin_prefetch(&V[base_bh + (long)(nb + NB + (tid >> 2)) * D], 0, 3);
    wait_tensor0();
    __syncthreads();

    // scores: 16 x 64 in 4 C-frags
    v8f sf[4] = {};
#pragma unroll
    for (int nf = 0; nf < 4; ++nf)
#pragma unroll
      for (int kk = 0; kk < 4; ++kk) {
        const __bf16* p = &Ks[nf * 16 + m][kk * 32 + kb];
        sf[nf] = wmma_bf16(qf[kk], load16(p, p + 8), sf[nf]);
      }
#pragma unroll
    for (int nf = 0; nf < 4; ++nf) {
      float mk = mask[(long)b * S + nb + nf * 16 + m];
#pragma unroll
      for (int e = 0; e < 8; ++e) sf[nf][e] = sf[nf][e] * scale + mk;
    }

    // online softmax; masks 1..8 keep the two 16-lane row halves separate
    float bmax[8];
#pragma unroll
    for (int r = 0; r < 8; ++r)
      bmax[r] = fmaxf(fmaxf(sf[0][r], sf[1][r]), fmaxf(sf[2][r], sf[3][r]));
#pragma unroll
    for (int off = 1; off < 16; off <<= 1)
#pragma unroll
      for (int r = 0; r < 8; ++r)
        bmax[r] = fmaxf(bmax[r], __shfl_xor(bmax[r], off, 32));
    float lf[8], rs[8];
#pragma unroll
    for (int r = 0; r < 8; ++r) {
      float mn = fmaxf(mrow[r], bmax[r]);
      lf[r] = __expf(mrow[r] - mn);
      mrow[r] = mn;
      rs[r] = 0.f;
    }
#pragma unroll
    for (int nf = 0; nf < 4; ++nf)
#pragma unroll
      for (int r = 0; r < 8; ++r) {
        float pv = __expf(sf[nf][r] - mrow[r]);
        sf[nf][r] = pv;
        rs[r] += pv;
      }
#pragma unroll
    for (int off = 1; off < 16; off <<= 1)
#pragma unroll
      for (int r = 0; r < 8; ++r) rs[r] += __shfl_xor(rs[r], off, 32);
#pragma unroll
    for (int r = 0; r < 8; ++r) lsum[r] = lsum[r] * lf[r] + rs[r];
#pragma unroll
    for (int df = 0; df < 8; ++df)
#pragma unroll
      for (int r = 0; r < 8; ++r) o[df][r] *= lf[r];

    // P: C-layout -> LDS -> A-layout
#pragma unroll
    for (int nf = 0; nf < 4; ++nf)
#pragma unroll
      for (int r = 0; r < 8; ++r)
        Ps[wave][r + rbase][nf * 16 + m] = f2bf(sf[nf][r]);
    __syncthreads();

    // O += P(16x64) x V(64x128)
#pragma unroll
    for (int kk2 = 0; kk2 < 2; ++kk2) {
      const __bf16* pp = &Ps[wave][m][kk2 * 32 + koff];
      v16bf pa = load16(pp, pp + 8);
#pragma unroll
      for (int df = 0; df < 8; ++df) {
        const __bf16* vp = &Vs[df * 16 + m][kk2 * 32 + kb];
        o[df] = wmma_bf16(pa, load16(vp, vp + 8), o[df]);
      }
    }
  }

#pragma unroll
  for (int df = 0; df < 8; ++df)
#pragma unroll
    for (int r = 0; r < 8; ++r) {
      long row = qbase + r + rbase;
      float v = o[df][r] / lsum[r];
      O[base_bh + row * (long)D + df * 16 + m] = f2bf(v);
    }
}

// ---------------------------------------------------------------- launch
extern "C" void kernel_launch(void* const* d_in, const int* in_sizes, int n_in,
                              void* d_out, int out_size, void* d_ws, size_t ws_size,
                              hipStream_t stream) {
  (void)in_sizes; (void)n_in; (void)out_size; (void)ws_size;
  const float* x  = (const float*)d_in[0];
  const float* fc = (const float*)d_in[1];
  const float* fs = (const float*)d_in[2];
  const float* mk = (const float*)d_in[3];
  const float* wq = (const float*)d_in[4];
  const float* wk = (const float*)d_in[5];
  const float* wv = (const float*)d_in[6];
  const float* wo = (const float*)d_in[7];
  float* out = (float*)d_out;

  const long MD = 8192L * 2048L;
  const long DD = 2048L * 2048L;

  char* w = (char*)d_ws;  // needs >= 192 MiB
  __bf16* xh  = (__bf16*)(w);
  __bf16* qh  = (__bf16*)(w + MD * 2);
  __bf16* kh  = (__bf16*)(w + MD * 4);
  __bf16* vh  = (__bf16*)(w + MD * 6);
  __bf16* oh  = (__bf16*)(w + MD * 8);
  __bf16* wqh = (__bf16*)(w + MD * 10);
  __bf16* wkh = (__bf16*)(w + MD * 10 + DD * 2);
  __bf16* wvh = (__bf16*)(w + MD * 10 + DD * 4);
  __bf16* woh = (__bf16*)(w + MD * 10 + DD * 6);

  cvt_f32_to_bf16<<<(int)(MD / 4 / 256), 256, 0, stream>>>(x,  xh,  MD / 4);
  cvt_f32_to_bf16<<<(int)(DD / 4 / 256), 256, 0, stream>>>(wq, wqh, DD / 4);
  cvt_f32_to_bf16<<<(int)(DD / 4 / 256), 256, 0, stream>>>(wk, wkh, DD / 4);
  cvt_f32_to_bf16<<<(int)(DD / 4 / 256), 256, 0, stream>>>(wv, wvh, DD / 4);
  cvt_f32_to_bf16<<<(int)(DD / 4 / 256), 256, 0, stream>>>(wo, woh, DD / 4);

  dim3 gg(2048 / 128, 8192 / 128);  // (16, 64)
  gemm_bf16_k<true ><<<gg, 128, 0, stream>>>(xh, wqh, qh, 8192, 2048, 2048);
  gemm_bf16_k<true ><<<gg, 128, 0, stream>>>(xh, wkh, kh, 8192, 2048, 2048);
  gemm_bf16_k<true ><<<gg, 128, 0, stream>>>(xh, wvh, vh, 8192, 2048, 2048);

  rope_k<<<(int)(8388608L / 256), 256, 0, stream>>>(qh, kh, fc, fs);

  flash_attn_k<<<dim3(64, 16), 256, 0, stream>>>(qh, kh, vh, mk, oh);

  gemm_bf16_k<false><<<gg, 128, 0, stream>>>(oh, woh, out, 8192, 2048, 2048);
}
